// MultiHeadAttention_36283883717392
// MI455X (gfx1250) — matrix operand order repaired and checked by reading, not compiled, not run on hardware
//
#include <hip/hip_runtime.h>

#define N_B 8
#define L_S 1024
#define E_D 512
#define H_N 8
#define D_H 64
#define NH_ (N_B * H_N)
#define VTS 40   // padded LDS stride (halves) for transposed V; 80B = 16B multiple

typedef __attribute__((ext_vector_type(16))) __bf16 v16bf;
typedef __attribute__((ext_vector_type(8)))  __bf16 v8bf;
typedef __attribute__((ext_vector_type(8)))  float  v8f;

union Frag { v16bf v; v8bf h[2]; uint4 u[2]; unsigned short s[16]; };

// native bf16 convert (lets the backend emit v_cvt_*bf16_f32 instead of bit math)
static __device__ __forceinline__ unsigned short bfbits(float f) {
  union { __bf16 h; unsigned short s; } c; c.h = (__bf16)f;
  return c.s;
}

// load 8 contiguous fp32 (32B-aligned) and convert to 8 bf16 (packed converts)
static __device__ __forceinline__ v8bf cvt8(const float* p) {
  return __builtin_convertvector(*(const v8f*)p, v8bf);
}

static __device__ __forceinline__ v8f zero8() {
  v8f z;
#pragma unroll
  for (int i = 0; i < 8; ++i) z[i] = 0.f;
  return z;
}

static __device__ __forceinline__ v8f wmma_bf16(const Frag& a, const Frag& b, v8f c) {
  return __builtin_amdgcn_wmma_f32_16x16x32_bf16(false, a.v, false, b.v, (short)0, c, false, false);
}

// ---------------------------------------------------------------------------
// Kernel 1: per-head projections  y[n,h,l,:] = x[n,l,h*64:h*64+64] @ W^T (bf16 out)
// ---------------------------------------------------------------------------
__global__ void __launch_bounds__(128)
proj_qkv_kernel(const float* __restrict__ xq, const float* __restrict__ xk,
                const float* __restrict__ xv,
                const float* __restrict__ Wq, const float* __restrict__ Wk,
                const float* __restrict__ Wv,
                unsigned short* __restrict__ yq, unsigned short* __restrict__ yk,
                unsigned short* __restrict__ yv)
{
  const float* x; const float* W; unsigned short* y;
  if (blockIdx.y == 0)      { x = xq; W = Wq; y = yq; }
  else if (blockIdx.y == 1) { x = xk; W = Wk; y = yk; }
  else                      { x = xv; W = Wv; y = yv; }

  const int tid  = threadIdx.x;
  const int lane = tid & 31, wave = tid >> 5;
  const int lh = lane >> 4, ln = lane & 15;

  const int bx   = blockIdx.x;
  const int nh   = bx >> 4;           // 16 row-blocks per (n,h)
  const int rb   = bx & 15;
  const int n    = nh >> 3, h = nh & 7;
  const int row0 = rb * 64 + wave * 16;

  // A fragments: two 16x32 bf16 tiles of the head slice (contraction d = 0..63)
  const float* xrow = x + ((size_t)n * L_S + row0 + ln) * E_D + h * D_H;
  Frag a0, a1;
  a0.h[0] = cvt8(xrow +  0 + 8 * lh);
  a0.h[1] = cvt8(xrow + 16 + 8 * lh);
  a1.h[0] = cvt8(xrow + 32 + 8 * lh);
  a1.h[1] = cvt8(xrow + 48 + 8 * lh);

  v8f acc[4];
#pragma unroll
  for (int t = 0; t < 4; ++t) acc[t] = zero8();

#pragma unroll
  for (int ot = 0; ot < 4; ++ot) {
    // B[d][o] = W[o][d]  -> lane (col o) reads a contiguous row of W
    const float* Wrow = W + (size_t)(ot * 16 + ln) * D_H;
    Frag b0, b1;
    b0.h[0] = cvt8(Wrow +  0 + 8 * lh);
    b0.h[1] = cvt8(Wrow + 16 + 8 * lh);
    b1.h[0] = cvt8(Wrow + 32 + 8 * lh);
    b1.h[1] = cvt8(Wrow + 48 + 8 * lh);
    acc[ot] = wmma_bf16(a0, b0, acc[ot]);
    acc[ot] = wmma_bf16(a1, b1, acc[ot]);
  }

#pragma unroll
  for (int ot = 0; ot < 4; ++ot)
#pragma unroll
    for (int vv = 0; vv < 8; ++vv) {
      int r = row0 + vv + 8 * lh;
      y[((size_t)nh * L_S + r) * D_H + ot * 16 + ln] = bfbits(acc[ot][vv]);
    }
}

// ---------------------------------------------------------------------------
// Kernel 2: flash attention per (n,h). 4 waves x 16 q-rows per block.
// K tile staged via CDNA5 async global->LDS (ASYNCcnt); V staged transposed.
// ---------------------------------------------------------------------------
__global__ void __launch_bounds__(128)
attn_kernel(const unsigned short* __restrict__ qbuf,
            const unsigned short* __restrict__ kbuf,
            const unsigned short* __restrict__ vbuf,
            const float* __restrict__ gmask,
            const unsigned char* __restrict__ pmask,
            unsigned short* __restrict__ att)
{
  __shared__ __align__(16) unsigned short Kb_s[32 * 64];        // K tile row-major
  __shared__ __align__(16) unsigned short VT_s[64 * VTS];       // V tile transposed
  __shared__ __align__(16) unsigned short Pb_s[4][16 * 32];     // per-wave P staging

  const int tid  = threadIdx.x;
  const int lane = tid & 31, wave = tid >> 5;
  const int lh = lane >> 4, ln = lane & 15;

  const int bx = blockIdx.x;
  const int nh = bx >> 4, qblk = bx & 15;
  const int n = nh >> 3, h = nh & 7;
  const int q0 = qblk * 64 + wave * 16;

  // Q A-fragments, fixed for the whole K loop
  const unsigned short* qrow = qbuf + ((size_t)nh * L_S + q0 + ln) * D_H;
  Frag qa0, qa1;
  qa0.u[0] = *(const uint4*)(qrow +  0 + 8 * lh);
  qa0.u[1] = *(const uint4*)(qrow + 16 + 8 * lh);
  qa1.u[0] = *(const uint4*)(qrow + 32 + 8 * lh);
  qa1.u[1] = *(const uint4*)(qrow + 48 + 8 * lh);

  float pq[8], mrun[8], lrun[8];
  v8f accO[4];
#pragma unroll
  for (int vv = 0; vv < 8; ++vv) {
    pq[vv]   = pmask[(size_t)n * L_S + q0 + vv + 8 * lh] ? 1.f : 0.f;
    mrun[vv] = -1e30f;
    lrun[vv] = 0.f;
  }
#pragma unroll
  for (int t = 0; t < 4; ++t) accO[t] = zero8();

  const float invs = 0.044194173824159216f;  // 1/sqrt(E=512)

  // per-thread staging coordinates
  const int kj = tid >> 2, kc = (tid & 3) * 16;    // K: 32 rows x 4 chunks of 16 halves
  const int vj = tid & 31, vdg = tid >> 5;         // V: 32 rows x 4 d-groups of 16

  for (int k0 = 0; k0 < L_S; k0 += 32) {
    __syncthreads();  // previous iteration's consumers done
    {   // stage K rows [k0, k0+32) x 64 halves, row-major — async copy to LDS
      unsigned ldsoff = (unsigned)(size_t)(&Kb_s[kj * 64 + kc]);
      const unsigned short* gsrc =
          kbuf + ((size_t)nh * L_S + k0 + kj) * D_H + kc;
      asm volatile(
          "global_load_async_to_lds_b128 %0, %1, off\n\t"
          "global_load_async_to_lds_b128 %0, %1, off offset:16"
          :: "v"(ldsoff), "v"(gsrc) : "memory");
    }
    {   // stage V rows transposed: VT[d][j]
      Frag t;
      const uint4* src =
          (const uint4*)(vbuf + ((size_t)nh * L_S + k0 + vj) * D_H + vdg * 16);
      t.u[0] = src[0]; t.u[1] = src[1];
#pragma unroll
      for (int i = 0; i < 16; ++i)
        VT_s[(vdg * 16 + i) * VTS + vj] = t.s[i];
    }
    asm volatile("s_wait_asynccnt 0" ::: "memory");
    __syncthreads();

    // S = Q K^T : two 16-col tiles, contraction d=64 (2 WMMAs each)
    float sc[2][8];
#pragma unroll
    for (int st = 0; st < 2; ++st) {
      const unsigned short* kr = &Kb_s[(st * 16 + ln) * 64];
      Frag bk0, bk1;
      bk0.u[0] = *(const uint4*)(kr + 8 * lh);
      bk0.u[1] = *(const uint4*)(kr + 16 + 8 * lh);
      bk1.u[0] = *(const uint4*)(kr + 32 + 8 * lh);
      bk1.u[1] = *(const uint4*)(kr + 48 + 8 * lh);
      v8f s = zero8();
      s = wmma_bf16(qa0, bk0, s);
      s = wmma_bf16(qa1, bk1, s);

      int kg = k0 + st * 16 + ln;
      float pk = pmask[(size_t)n * L_S + kg] ? 1.f : 0.f;
      const float* gmr = gmask + (size_t)n * L_S * L_S + kg;
#pragma unroll
      for (int vv = 0; vv < 8; ++vv) {
        int qg = q0 + vv + 8 * lh;
        float e = s[vv] * gmr[(size_t)qg * L_S];       // multiplicative graph mask
        e = (pq[vv] * pk != 0.f) ? e : -1e20f;         // pad outer-product mask
        sc[st][vv] = e * invs;
      }
    }

    // online softmax across the 32 staged columns (rows live in 16-lane halves)
#pragma unroll
    for (int vv = 0; vv < 8; ++vv) {
      float m2 = fmaxf(sc[0][vv], sc[1][vv]);
#pragma unroll
      for (int off = 1; off < 16; off <<= 1)
        m2 = fmaxf(m2, __shfl_xor(m2, off, 32));
      float mnew = fmaxf(mrun[vv], m2);
      float corr = __expf(mrun[vv] - mnew);
      float p0 = __expf(sc[0][vv] - mnew);
      float p1 = __expf(sc[1][vv] - mnew);
      float ls = p0 + p1;
#pragma unroll
      for (int off = 1; off < 16; off <<= 1)
        ls += __shfl_xor(ls, off, 32);
      lrun[vv] = lrun[vv] * corr + ls;
      mrun[vv] = mnew;
#pragma unroll
      for (int dt = 0; dt < 4; ++dt) accO[dt][vv] *= corr;
      // C-layout -> LDS (row-major), read back in A layout below
      Pb_s[wave][(vv + 8 * lh) * 32 + ln]      = bfbits(p0);
      Pb_s[wave][(vv + 8 * lh) * 32 + 16 + ln] = bfbits(p1);
    }

    Frag pa;
    const unsigned short* pr = &Pb_s[wave][ln * 32];
    pa.u[0] = *(const uint4*)(pr +  0 + 8 * lh);
    pa.u[1] = *(const uint4*)(pr + 16 + 8 * lh);

    // O += P V
#pragma unroll
    for (int dt = 0; dt < 4; ++dt) {
      const unsigned short* vr = &VT_s[(dt * 16 + ln) * VTS];
      Frag bv;
      bv.u[0] = *(const uint4*)(vr + 8 * lh);
      bv.u[1] = *(const uint4*)(vr + 16 + 8 * lh);
      accO[dt] = wmma_bf16(pa, bv, accO[dt]);
    }
  }

  // normalize and store concat-head bf16 [n][l][h*64+d]
#pragma unroll
  for (int dt = 0; dt < 4; ++dt)
#pragma unroll
    for (int vv = 0; vv < 8; ++vv) {
      int qg = q0 + vv + 8 * lh;
      att[((size_t)n * L_S + qg) * E_D + h * D_H + dt * 16 + ln] =
          bfbits(accO[dt][vv] / lrun[vv]);
    }
}

// ---------------------------------------------------------------------------
// Kernel 3: out = att @ Wo^T + bo   (8192x512 @ 512x512, fp32 out)
// ---------------------------------------------------------------------------
__global__ void __launch_bounds__(128)
outproj_kernel(const unsigned short* __restrict__ att, const float* __restrict__ Wo,
               const float* __restrict__ bo, float* __restrict__ out)
{
  const int tid  = threadIdx.x;
  const int lane = tid & 31, wave = tid >> 5;
  const int lh = lane >> 4, ln = lane & 15;
  const int r0 = blockIdx.x * 64 + wave * 16;
  const int o0 = blockIdx.y * 64;

  v8f acc[4];
#pragma unroll
  for (int t = 0; t < 4; ++t) acc[t] = zero8();

  const unsigned short* arow = att + (size_t)(r0 + ln) * E_D;

  for (int c0 = 0; c0 < E_D; c0 += 32) {
    Frag a;
    a.u[0] = *(const uint4*)(arow + c0 +  0 + 8 * lh);
    a.u[1] = *(const uint4*)(arow + c0 + 16 + 8 * lh);
#pragma unroll
    for (int ot = 0; ot < 4; ++ot) {
      // B[c][o] = Wo[o][c] -> per-lane contiguous row of Wo
      const float* wrow = Wo + (size_t)(o0 + ot * 16 + ln) * E_D + c0 + 8 * lh;
      Frag b;
      b.h[0] = cvt8(wrow);
      b.h[1] = cvt8(wrow + 16);
      acc[ot] = wmma_bf16(a, b, acc[ot]);
    }
  }

#pragma unroll
  for (int ot = 0; ot < 4; ++ot) {
    int o = o0 + ot * 16 + ln;
    float bb = bo[o];
#pragma unroll
    for (int vv = 0; vv < 8; ++vv)
      out[(size_t)(r0 + vv + 8 * lh) * E_D + o] = acc[ot][vv] + bb;
  }
}

// ---------------------------------------------------------------------------
extern "C" void kernel_launch(void* const* d_in, const int* in_sizes, int n_in,
                              void* d_out, int out_size, void* d_ws, size_t ws_size,
                              hipStream_t stream)
{
  (void)in_sizes; (void)n_in; (void)out_size; (void)ws_size;
  const float*         values = (const float*)d_in[0];
  const float*         keys   = (const float*)d_in[1];
  const float*         query  = (const float*)d_in[2];
  const unsigned char* pad    = (const unsigned char*)d_in[3];
  const float*         gmask  = (const float*)d_in[4];
  const float*         Wv     = (const float*)d_in[5];
  const float*         Wk     = (const float*)d_in[6];
  const float*         Wq     = (const float*)d_in[7];
  const float*         Wo     = (const float*)d_in[8];
  const float*         bo     = (const float*)d_in[9];

  const size_t perbuf = (size_t)NH_ * L_S * D_H;      // halves (8 MB each)
  unsigned short* qbuf = (unsigned short*)d_ws;
  unsigned short* kbuf = qbuf + perbuf;
  unsigned short* vbuf = kbuf + perbuf;
  unsigned short* atts = vbuf + perbuf;

  proj_qkv_kernel<<<dim3(NH_ * (L_S / 64), 3), 128, 0, stream>>>(
      query, keys, values, Wq, Wk, Wv, qbuf, kbuf, vbuf);
  attn_kernel<<<dim3(NH_ * (L_S / 64)), 128, 0, stream>>>(
      qbuf, kbuf, vbuf, gmask, pad, atts);
  outproj_kernel<<<dim3((N_B * L_S) / 64, E_D / 64), 128, 0, stream>>>(
      atts, Wo, bo, (float*)d_out);
}
